// MPC_31052613550455
// MI455X (gfx1250) — matrix operand order repaired and checked by reading, not compiled, not run on hardware
//
#include <hip/hip_runtime.h>
#include <hip/hip_bf16.h>
#include <math.h>

// ---------------- problem constants (from reference) ----------------
#define C_    192
#define NPIX  16384     // h*w = n tokens per batch
#define NB    8
#define NGCOL 256       // ng (conv width)
#define GSROW 64        // gs
#define KP    200       // padded LDS row stride (bf16 elems): 400B, 16B aligned, conflict-free

typedef __attribute__((ext_vector_type(16))) __bf16 v16bf;
typedef __attribute__((ext_vector_type(8)))  __bf16 v8bf;
typedef __attribute__((ext_vector_type(8)))  float  v8f;

// Build a 16-element bf16 A/B fragment from two 16-byte chunks.
__device__ __forceinline__ v16bf ldpair(const __bf16* p0, const __bf16* p1) {
    v8bf lo = *(const v8bf*)p0;
    v8bf hi = *(const v8bf*)p1;
    v16bf r;
#pragma unroll
    for (int e = 0; e < 8; ++e) { r[e] = lo[e]; r[8 + e] = hi[e]; }
    return r;
}

// D[mbase..mbase+32) x N=64  +=  W[192x192, row-major] * Xs^T   (Xs stored [n][k])
// K = 192 in 6 steps of 32. Per CDNA5 16-bit WMMA layouts:
//  A (16x32): lane 0-15 -> M=lane, K elems {kb+0..7, kb+16..23}; lane 16-31 -> {kb+8..15, kb+24..31}
//  B (32x16): lane 0-15 -> N=lane, K elems {kb+0..7, kb+16..23}; lane 16-31 -> {kb+8..15, kb+24..31}
//  C/D      : vgpr r -> M = r (+8 for lanes 16-31), N = lane&15
__device__ __forceinline__ void gemm_192x64(const __bf16* __restrict__ W,
                                            const __bf16 (*Xs)[KP],
                                            int mbase, int lane, v8f acc[2][4]) {
    const int lm     = lane & 15;
    const int khalfA = (lane < 16) ? 0 : 8;
    const int khalfB = (lane < 16) ? 0 : 8;
#pragma unroll
    for (int ks = 0; ks < 6; ++ks) {
        const int kb = ks * 32;
        const __bf16* ar0 = W + (size_t)(mbase      + lm) * C_ + kb + khalfA;
        const __bf16* ar1 = W + (size_t)(mbase + 16 + lm) * C_ + kb + khalfA;
        v16bf a0 = ldpair(ar0, ar0 + 16);
        v16bf a1 = ldpair(ar1, ar1 + 16);
#pragma unroll
        for (int nt = 0; nt < 4; ++nt) {
            const __bf16* bp = &Xs[nt * 16 + lm][kb + khalfB];
            v16bf bf = ldpair(bp, bp + 16);
            acc[0][nt] = __builtin_amdgcn_wmma_f32_16x16x32_bf16(
                false, a0, false, bf, (short)0, acc[0][nt], false, false);
            acc[1][nt] = __builtin_amdgcn_wmma_f32_16x16x32_bf16(
                false, a1, false, bf, (short)0, acc[1][nt], false, false);
        }
    }
}

// ---------------- kernel 0: fp32 -> bf16 weight prep ----------------
__global__ void wprep(const float* __restrict__ w1, const float* __restrict__ wl1,
                      const float* __restrict__ w2,
                      __bf16* __restrict__ wb1, __bf16* __restrict__ wbl1,
                      __bf16* __restrict__ wb2) {
    int i = blockIdx.x * 256 + threadIdx.x;
    if (i < C_ * C_) {
        wb1[i]  = (__bf16)w1[i];
        wbl1[i] = (__bf16)wl1[i];
        wb2[i]  = (__bf16)w2[i];
    }
}

// ---------------- kernel 1: per-batch stable ascending bitonic argsort ---------
__global__ __launch_bounds__(1024) void sortk(const float* __restrict__ sa,
                                              int* __restrict__ idxg) {
    extern __shared__ unsigned char smem[];
    float* key = (float*)smem;
    int*   sid = (int*)(smem + (size_t)NPIX * sizeof(float));
    const int tid = threadIdx.x;
    const int b   = blockIdx.x;
    for (int i = tid; i < NPIX; i += 1024) { key[i] = sa[b * NPIX + i]; sid[i] = i; }
    __syncthreads();
    for (int k = 2; k <= NPIX; k <<= 1) {
        for (int j = k >> 1; j > 0; j >>= 1) {
            for (int i = tid; i < NPIX; i += 1024) {
                int ixj = i ^ j;
                if (ixj > i) {
                    float ka = key[i], kb = key[ixj];
                    int   ia = sid[i], ib = sid[ixj];
                    bool gt = (ka > kb) || (ka == kb && ia > ib);   // stable tiebreak
                    bool lt = (ka < kb) || (ka == kb && ia < ib);
                    bool doswap = ((i & k) == 0) ? gt : lt;
                    if (doswap) {
                        key[i] = kb; key[ixj] = ka;
                        sid[i] = ib; sid[ixj] = ia;
                    }
                }
            }
            __syncthreads();
        }
    }
    for (int i = tid; i < NPIX; i += 1024) idxg[b * NPIX + i] = sid[i];
}

// ---------------- kernel 2: gather + GEMM(w1) + bias + erf-GELU -> y (bf16) ----
__global__ __launch_bounds__(192) void gemm1(const float* __restrict__ x,
                                             const int* __restrict__ idxg,
                                             const __bf16* __restrict__ wb1,
                                             const float* __restrict__ b1,
                                             __bf16* __restrict__ yws) {
    __shared__ __bf16 Xs[64][KP];
    __shared__ int    idxs[64];
    const int tid = threadIdx.x;
    const int b   = blockIdx.y;
    const int t0  = blockIdx.x * 64;
    if (tid < 64) idxs[tid] = idxg[b * NPIX + t0 + tid];
    __syncthreads();
    {   // thread = channel k, gather 64 tokens (L2-resident per-channel planes)
        const float* xp = x + ((size_t)(b * C_) + tid) * NPIX;
#pragma unroll 4
        for (int n = 0; n < 64; ++n) Xs[n][tid] = (__bf16)xp[idxs[n]];
    }
    __syncthreads();

    const int lane = tid & 31, wave = tid >> 5;
    const int mbase = wave * 32, lm = lane & 15;
    v8f z = {0.f, 0.f, 0.f, 0.f, 0.f, 0.f, 0.f, 0.f};
    v8f acc[2][4] = {{z, z, z, z}, {z, z, z, z}};
    gemm_192x64(wb1, Xs, mbase, lane, acc);

#pragma unroll
    for (int mt = 0; mt < 2; ++mt)
#pragma unroll
        for (int nt = 0; nt < 4; ++nt)
#pragma unroll
            for (int r = 0; r < 8; ++r) {
                int m = mbase + mt * 16 + r + ((lane >= 16) ? 8 : 0);
                int n = nt * 16 + lm;
                float v = acc[mt][nt][r] + b1[m];
                v = 0.5f * v * (1.0f + erff(v * 0.70710678118654752f)); // exact GELU
                yws[((size_t)(b * C_) + m) * NPIX + t0 + n] = (__bf16)v;
            }
}

// ---------------- kernel 3: fused depthwise (1x5 d1) then (1x5 d3) conv -------
// y layout [b][c][t], t = j*64 + g ; conv along j (stride 64 in t), zero-pad exact.
__global__ __launch_bounds__(256) void dwconv(const __bf16* __restrict__ yws,
                                              const float* __restrict__ wc0,
                                              const float* __restrict__ bc0,
                                              const float* __restrict__ wsp,
                                              const float* __restrict__ bsp,
                                              __bf16* __restrict__ attnws) {
    const int c = blockIdx.y, b = blockIdx.z;
    const int t = blockIdx.x * 256 + threadIdx.x;
    const int j = t >> 6, g = t & 63;
    const __bf16* yp = yws + ((size_t)(b * C_) + c) * NPIX + g;
    float w0[5], w3[5];
#pragma unroll
    for (int i = 0; i < 5; ++i) { w0[i] = wc0[c * 5 + i]; w3[i] = wsp[c * 5 + i]; }
    const float bias0 = bc0[c];
    float a2 = bsp[c];
#pragma unroll
    for (int q = 0; q < 5; ++q) {
        int jq = j + 3 * (q - 2);
        if (jq >= 0 && jq < NGCOL) {          // second conv zero-pads attn1 outside range
            float a1 = bias0;
#pragma unroll
            for (int p = 0; p < 5; ++p) {
                int jp = jq + (p - 2);
                if (jp >= 0 && jp < NGCOL) a1 += w0[p] * (float)yp[jp * 64];
            }
            a2 += w3[q] * a1;
        }
    }
    attnws[((size_t)(b * C_) + c) * NPIX + t] = (__bf16)a2;
}

// ------- kernel 4: GEMM(wl1)+bias -> gate with y -> GEMM(w2)+bias -> scatter+residual
__global__ __launch_bounds__(192) void gemm2(const float* __restrict__ x,
                                             const int* __restrict__ idxg,
                                             const __bf16* __restrict__ yws,
                                             const __bf16* __restrict__ attnws,
                                             const __bf16* __restrict__ wbl1,
                                             const float* __restrict__ bl1,
                                             const __bf16* __restrict__ wb2,
                                             const float* __restrict__ b2,
                                             float* __restrict__ out) {
    __shared__ __bf16 As[64][KP];   // attn2 tile
    __shared__ __bf16 Gs[64][KP];   // gated tile
    __shared__ int    idxs[64];
    const int tid = threadIdx.x;
    const int b   = blockIdx.y;
    const int t0  = blockIdx.x * 64;
    if (tid < 64) idxs[tid] = idxg[b * NPIX + t0 + tid];
    for (int e = tid; e < C_ * 64; e += 192) {
        int n = e & 63, k = e >> 6;
        As[n][k] = attnws[((size_t)(b * C_) + k) * NPIX + t0 + n];
    }
    __syncthreads();

    const int lane = tid & 31, wave = tid >> 5;
    const int mbase = wave * 32, lm = lane & 15;
    v8f z = {0.f, 0.f, 0.f, 0.f, 0.f, 0.f, 0.f, 0.f};

    v8f acc[2][4] = {{z, z, z, z}, {z, z, z, z}};
    gemm_192x64(wbl1, As, mbase, lane, acc);
#pragma unroll
    for (int mt = 0; mt < 2; ++mt)
#pragma unroll
        for (int nt = 0; nt < 4; ++nt)
#pragma unroll
            for (int r = 0; r < 8; ++r) {
                int m = mbase + mt * 16 + r + ((lane >= 16) ? 8 : 0);
                int n = nt * 16 + lm;
                float av = acc[mt][nt][r] + bl1[m];
                float uv = (float)yws[((size_t)(b * C_) + m) * NPIX + t0 + n];
                Gs[n][m] = (__bf16)(uv * av);
            }
    __syncthreads();

    v8f acc2[2][4] = {{z, z, z, z}, {z, z, z, z}};
    gemm_192x64(wb2, Gs, mbase, lane, acc2);
#pragma unroll
    for (int mt = 0; mt < 2; ++mt)
#pragma unroll
        for (int nt = 0; nt < 4; ++nt)
#pragma unroll
            for (int r = 0; r < 8; ++r) {
                int m = mbase + mt * 16 + r + ((lane >= 16) ? 8 : 0);
                int n = nt * 16 + lm;
                int p = idxs[n];                         // scatter back to pixel order
                size_t o = ((size_t)(b * C_) + m) * NPIX + p;
                out[o] = x[o] + acc2[mt][nt][r] + b2[m]; // residual folded in
            }
}

// ---------------- host launcher ----------------
extern "C" void kernel_launch(void* const* d_in, const int* in_sizes, int n_in,
                              void* d_out, int out_size, void* d_ws, size_t ws_size,
                              hipStream_t stream) {
    (void)in_sizes; (void)n_in; (void)out_size; (void)ws_size;
    const float* x   = (const float*)d_in[0];
    const float* sa  = (const float*)d_in[1];
    const float* w1  = (const float*)d_in[2];
    const float* b1  = (const float*)d_in[3];
    const float* wc0 = (const float*)d_in[4];
    const float* bc0 = (const float*)d_in[5];
    const float* wsp = (const float*)d_in[6];
    const float* bsp = (const float*)d_in[7];
    const float* wl1 = (const float*)d_in[8];
    const float* bl1 = (const float*)d_in[9];
    const float* w2  = (const float*)d_in[10];
    const float* b2  = (const float*)d_in[11];
    float* out = (float*)d_out;

    unsigned char* ws = (unsigned char*)d_ws;
    size_t off = 0;
    int*    idxg = (int*)(ws + off);    off += (size_t)NB * NPIX * sizeof(int);   // 512 KB
    __bf16* wb1  = (__bf16*)(ws + off); off += (size_t)C_ * C_ * 2;
    __bf16* wbl1 = (__bf16*)(ws + off); off += (size_t)C_ * C_ * 2;
    __bf16* wb2  = (__bf16*)(ws + off); off += (size_t)C_ * C_ * 2;
    off = (off + 255) & ~(size_t)255;
    __bf16* yws    = (__bf16*)(ws + off); off += (size_t)NB * C_ * NPIX * 2;      // 50.3 MB
    __bf16* attnws = (__bf16*)(ws + off); off += (size_t)NB * C_ * NPIX * 2;      // 50.3 MB

    wprep<<<(C_ * C_ + 255) / 256, 256, 0, stream>>>(w1, wl1, w2, wb1, wbl1, wb2);
    sortk<<<NB, 1024, (size_t)NPIX * 8, stream>>>(sa, idxg);
    gemm1<<<dim3(NPIX / 64, NB), 192, 0, stream>>>(x, idxg, wb1, b1, yws);
    dwconv<<<dim3(NPIX / 256, C_, NB), 256, 0, stream>>>(yws, wc0, bc0, wsp, bsp, attnws);
    gemm2<<<dim3(NPIX / 64, NB), 192, 0, stream>>>(x, idxg, yws, attnws, wbl1, bl1, wb2, b2, out);
}
